// DeformNet2_61134564491588
// MI455X (gfx1250) — compile-verified, matrix operand order checked by reading, not run on hardware
//
#include <hip/hip_runtime.h>
#include <hip/hip_bf16.h>
#include <math.h>

// ---------------- types for WMMA ----------------
typedef __bf16 v16bf __attribute__((ext_vector_type(16)));
typedef float  v8f   __attribute__((ext_vector_type(8)));

union FragU { v16bf v; unsigned int u[8]; };

// ---------------- constants ----------------
#define BATCH 256
#define HW    784          // 28*28
#define MTOT  (BATCH*HW)   // 200704 = 3136*64
#define KDIM  288          // 32*9
#define KDW   144          // 288 bf16 -> 144 dwords (payload)
#define ASTRIDE 148        // padded LDS row stride in dwords (bank-conflict free)

// ws layout (float units)
#define WS_H1   0
#define WS_OFF  6422528
#define WS_H2   10035200
#define WS_POOL 16457728
#define WS_WB   16474112   // bf16 weights: 32*288 (wp pad) + 32*288 (w2) + 64*288 (w3)

__device__ __forceinline__ unsigned short f2bf(float f) {
    unsigned int u = __float_as_uint(f);
    u += 0x7FFFu + ((u >> 16) & 1u);      // round-to-nearest-even
    return (unsigned short)(u >> 16);
}
__device__ __forceinline__ unsigned int pack2bf(float a, float b) {
    return (unsigned int)f2bf(a) | ((unsigned int)f2bf(b) << 16);
}

// Load one 16x32 bf16 fragment (A-layout; also used for B as Bt[n][k] rows).
// base: dword array, rowStrideDw dwords per row; row16: first of 16 rows;
// kb: K offset (multiple of 32). Groups u[0..3] and u[4..7] are contiguous
// 16B-aligned runs -> compiler merges to (ds_|global_)load_b128.
__device__ __forceinline__ v16bf frag16(const unsigned int* base, int rowStrideDw,
                                        int row16, int kb, int lane) {
    FragU f;
    int r = row16 + (lane & 15);
    int khalf = (lane & 16) ? 8 : 0;
#pragma unroll
    for (int i = 0; i < 8; ++i) {
        int k = khalf + ((i < 4) ? (2 * i) : (16 + 2 * (i - 4)));
        f.u[i] = base[r * rowStrideDw + ((kb + k) >> 1)];
    }
    return f.v;
}

__device__ __forceinline__ v8f wmma_bf16(v16bf a, v16bf b, v8f c) {
    return __builtin_amdgcn_wmma_f32_16x16x32_bf16(false, a, false, b, (short)0, c,
                                                   false, false);
}

// im2col read with zero padding, feature map (B,C,28,28), k = c*9 + tap
__device__ __forceinline__ float im2col_h(const float* __restrict__ h, int b, int C,
                                          int k, int y, int x) {
    int c = k / 9, tap = k % 9;
    int yy = y + tap / 3 - 1, xx = x + tap % 3 - 1;
    if (yy < 0 || yy >= 28 || xx < 0 || xx >= 28) return 0.f;
    return h[((b * C + c) * 28 + yy) * 28 + xx];
}

// ---------------- kernel 0: zero pooled accumulator ----------------
__global__ void zero_pool_kernel(float* __restrict__ pool) {
    int i = blockIdx.x * 256 + threadIdx.x;
    if (i < BATCH * 64) pool[i] = 0.f;
}

// ---------------- kernel 1: conv1(3->32) + relu + bn ----------------
__global__ void conv1_kernel(const float* __restrict__ x, const float* __restrict__ w1,
                             const float* __restrict__ g1, const float* __restrict__ b1,
                             const float* __restrict__ m1, const float* __restrict__ v1,
                             float* __restrict__ h1) {
    int idx = blockIdx.x * 256 + threadIdx.x;
    if (idx >= BATCH * 32 * HW) return;
    int pos = idx % HW;
    int oc  = (idx / HW) % 32;
    int b   = idx / (32 * HW);
    int y = pos / 28, x0 = pos % 28;
    float acc = 0.f;
#pragma unroll
    for (int c = 0; c < 3; ++c)
#pragma unroll
        for (int dy = 0; dy < 3; ++dy)
#pragma unroll
            for (int dx = 0; dx < 3; ++dx) {
                int yy = y + dy - 1, xx = x0 + dx - 1;
                if (yy >= 0 && yy < 28 && xx >= 0 && xx < 28)
                    acc += x[((b * 3 + c) * 28 + yy) * 28 + xx] *
                           w1[((oc * 3 + c) * 3 + dy) * 3 + dx];
            }
    float inv = g1[oc] * rsqrtf(v1[oc] + 1e-5f);
    h1[idx] = fmaxf(acc, 0.f) * inv + (b1[oc] - m1[oc] * inv);
}

// ---------------- kernel 2: fp32 weights -> padded bf16 ----------------
__global__ void convert_weights_kernel(const float* __restrict__ wp,
                                       const float* __restrict__ w2,
                                       const float* __restrict__ w3,
                                       unsigned short* __restrict__ wb) {
    int t = blockIdx.x * 256 + threadIdx.x;
    if (t < 32 * KDIM) {                       // wp: 18x288 padded to 32x288
        int n = t / KDIM, k = t % KDIM;
        wb[t] = (n < 18) ? f2bf(wp[n * KDIM + k]) : (unsigned short)0;
    } else if (t < 64 * KDIM) {                // w2: 32x288
        wb[t] = f2bf(w2[t - 32 * KDIM]);
    } else if (t < 128 * KDIM) {               // w3: 64x288
        wb[t] = f2bf(w3[t - 64 * KDIM]);
    }
}

// ---------------- kernel 3: p_conv (32->18) via WMMA GEMM ----------------
__global__ void __launch_bounds__(128)
pconv_wmma_kernel(const float* __restrict__ h1, const unsigned short* __restrict__ wpb,
                  const float* __restrict__ bp, float* __restrict__ offs) {
    __shared__ unsigned int ldsA[64 * ASTRIDE];    // ~37.9 KB
    const int m0 = blockIdx.x * 64;
    __builtin_prefetch(wpb + threadIdx.x * 64, 0, 1);   // global_prefetch_b8: warm B
    for (int t = threadIdx.x; t < 64 * KDW; t += 128) {
        int r = t / KDW, kdw = t % KDW;
        int m = m0 + r;
        int b = m / HW, pos = m % HW, y = pos / 28, x = pos % 28;
        ldsA[r * ASTRIDE + kdw] = pack2bf(im2col_h(h1, b, 32, 2 * kdw, y, x),
                                          im2col_h(h1, b, 32, 2 * kdw + 1, y, x));
    }
    __syncthreads();
    const int lane = threadIdx.x & 31;
    const int rowBase = (threadIdx.x >> 5) * 16;
    const unsigned int* Bg = (const unsigned int*)wpb;   // 32 rows x 144 dw
#pragma unroll
    for (int ct = 0; ct < 2; ++ct) {
        v8f acc = {};
#pragma unroll
        for (int kb = 0; kb < KDIM; kb += 32) {
            v16bf a  = frag16(ldsA, ASTRIDE, rowBase, kb, lane);
            v16bf bm = frag16(Bg,   KDW,     ct * 16, kb, lane);
            acc = wmma_bf16(a, bm, acc);
        }
        int n = ct * 16 + (lane & 15);
        if (n < 18) {
            float bias = bp[n];
            int rext = (lane & 16) ? 8 : 0;
#pragma unroll
            for (int r = 0; r < 8; ++r) {
                int m = m0 + rowBase + rext + r;
                int b = m / HW, pos = m % HW;
                offs[(b * 18 + n) * HW + pos] = acc[r] + bias;
            }
        }
    }
}

// ---------------- kernel 4: deformable conv (32->32) + relu + bn ----------------
struct Samp { short i0, i1, i2, i3; float w0, w1, w2, w3; };

__device__ __forceinline__ short pad_idx(float qy, float qx) {
    int yy = (int)qy - 1, xx = (int)qx - 1;   // padded(30x30) -> h1(28x28)
    if (yy < 0 || yy >= 28 || xx < 0 || xx >= 28) return (short)-1;
    return (short)(yy * 28 + xx);
}

__global__ void __launch_bounds__(128)
deform_wmma_kernel(const float* __restrict__ h1, const float* __restrict__ offs,
                   const unsigned short* __restrict__ w2b,
                   const float* __restrict__ g2, const float* __restrict__ b2,
                   const float* __restrict__ m2, const float* __restrict__ v2,
                   float* __restrict__ h2) {
    __shared__ unsigned int ldsA[64 * ASTRIDE];    // ~37.9 KB
    __shared__ Samp samp[64 * 9];                  // 13.8 KB
    const int m0 = blockIdx.x * 64;
    __builtin_prefetch(w2b + threadIdx.x * 64, 0, 1);   // global_prefetch_b8

    // Stage 1: sampling metadata (64 rows x 9 taps)
    for (int t = threadIdx.x; t < 64 * 9; t += 128) {
        int r = t / 9, tap = t % 9;
        int m = m0 + r;
        int b = m / HW, pos = m % HW, y = pos / 28, x = pos % 28;
        float oy = offs[(b * 18 + tap) * HW + pos];
        float ox = offs[(b * 18 + 9 + tap) * HW + pos];
        float py = (float)(y + tap / 3) + oy;          // (y+1) + (tap/3 - 1)
        float px = (float)(x + tap % 3) + ox;
        float fy = floorf(py), fx = floorf(px);
        float qy0 = fminf(fmaxf(fy, 0.f), 29.f);
        float qy1 = fminf(fmaxf(fy + 1.f, 0.f), 29.f);
        float qx0 = fminf(fmaxf(fx, 0.f), 29.f);
        float qx1 = fminf(fmaxf(fx + 1.f, 0.f), 29.f);
        py = fminf(fmaxf(py, 0.f), 29.f);
        px = fminf(fmaxf(px, 0.f), 29.f);
        Samp s;
        s.w0 = (1.f + (qy0 - py)) * (1.f + (qx0 - px));   // lt
        s.w1 = (1.f - (qy1 - py)) * (1.f - (qx1 - px));   // rb
        s.w2 = (1.f + (qy0 - py)) * (1.f - (qx1 - px));   // lb
        s.w3 = (1.f - (qy1 - py)) * (1.f + (qx0 - px));   // rt
        s.i0 = pad_idx(qy0, qx0);
        s.i1 = pad_idx(qy1, qx1);
        s.i2 = pad_idx(qy0, qx1);
        s.i3 = pad_idx(qy1, qx0);
        samp[t] = s;
    }
    __syncthreads();

    // Stage 2: bilinear gather -> bf16 A-tile (row m, k = c*9 + tap)
    for (int t = threadIdx.x; t < 64 * KDW; t += 128) {
        int r = t / KDW, kdw = t % KDW;
        int m = m0 + r, b = m / HW;
        float v[2];
#pragma unroll
        for (int e = 0; e < 2; ++e) {
            int k = 2 * kdw + e, c = k / 9, tap = k % 9;
            Samp s = samp[r * 9 + tap];
            const float* plane = h1 + (b * 32 + c) * HW;
            float a = 0.f;
            if (s.i0 >= 0) a += s.w0 * plane[s.i0];
            if (s.i1 >= 0) a += s.w1 * plane[s.i1];
            if (s.i2 >= 0) a += s.w2 * plane[s.i2];
            if (s.i3 >= 0) a += s.w3 * plane[s.i3];
            v[e] = a;
        }
        ldsA[r * ASTRIDE + kdw] = pack2bf(v[0], v[1]);
    }
    __syncthreads();

    // Stage 3: WMMA GEMM, epilogue relu + bn
    const int lane = threadIdx.x & 31;
    const int rowBase = (threadIdx.x >> 5) * 16;
    const unsigned int* Bg = (const unsigned int*)w2b;   // 32 x 144 dw
#pragma unroll
    for (int ct = 0; ct < 2; ++ct) {
        v8f acc = {};
#pragma unroll
        for (int kb = 0; kb < KDIM; kb += 32) {
            v16bf a  = frag16(ldsA, ASTRIDE, rowBase, kb, lane);
            v16bf bm = frag16(Bg,   KDW,     ct * 16, kb, lane);
            acc = wmma_bf16(a, bm, acc);
        }
        int n = ct * 16 + (lane & 15);
        float inv  = g2[n] * rsqrtf(v2[n] + 1e-5f);
        float beta = b2[n] - m2[n] * inv;
        int rext = (lane & 16) ? 8 : 0;
#pragma unroll
        for (int r = 0; r < 8; ++r) {
            int m = m0 + rowBase + rext + r;
            int b = m / HW, pos = m % HW;
            h2[(b * 32 + n) * HW + pos] = fmaxf(acc[r], 0.f) * inv + beta;
        }
    }
}

// ---------------- kernel 5: conv3 (32->64) + relu + bn + global mean ----------------
__global__ void __launch_bounds__(128)
conv3_wmma_kernel(const float* __restrict__ h2, const unsigned short* __restrict__ w3b,
                  const float* __restrict__ g3, const float* __restrict__ b3,
                  const float* __restrict__ m3, const float* __restrict__ v3,
                  float* __restrict__ pool) {
    __shared__ unsigned int ldsA[64 * ASTRIDE];    // ~37.9 KB
    __shared__ float poolAcc[2 * 64];              // per-block partial pool (<=2 images)
    const int m0 = blockIdx.x * 64;
    const int bFirst = m0 / HW;
    const int bLast  = (m0 + 63) / HW;
    __builtin_prefetch(w3b + threadIdx.x * 128, 0, 1);  // global_prefetch_b8

    poolAcc[threadIdx.x] = 0.f;                    // 128 threads == 128 slots
    for (int t = threadIdx.x; t < 64 * KDW; t += 128) {
        int r = t / KDW, kdw = t % KDW;
        int m = m0 + r;
        int b = m / HW, pos = m % HW, y = pos / 28, x = pos % 28;
        ldsA[r * ASTRIDE + kdw] = pack2bf(im2col_h(h2, b, 32, 2 * kdw, y, x),
                                          im2col_h(h2, b, 32, 2 * kdw + 1, y, x));
    }
    __syncthreads();
    const int lane = threadIdx.x & 31;
    const int rowBase = (threadIdx.x >> 5) * 16;
    const unsigned int* Bg = (const unsigned int*)w3b;   // 64 x 144 dw
#pragma unroll
    for (int ct = 0; ct < 4; ++ct) {
        v8f acc = {};
#pragma unroll
        for (int kb = 0; kb < KDIM; kb += 32) {
            v16bf a  = frag16(ldsA, ASTRIDE, rowBase, kb, lane);
            v16bf bm = frag16(Bg,   KDW,     ct * 16, kb, lane);
            acc = wmma_bf16(a, bm, acc);
        }
        int n = ct * 16 + (lane & 15);
        float inv  = g3[n] * rsqrtf(v3[n] + 1e-5f);
        float beta = b3[n] - m3[n] * inv;
        int rext = (lane & 16) ? 8 : 0;
        float s0 = 0.f, s1 = 0.f;                  // per-lane partials per image slot
#pragma unroll
        for (int r = 0; r < 8; ++r) {
            int m = m0 + rowBase + rext + r;
            int b = m / HW;
            float val = fmaxf(acc[r], 0.f) * inv + beta;
            if (b == bFirst) s0 += val; else s1 += val;
        }
        atomicAdd(&poolAcc[n], s0);                // ds_add_f32 (unconditional: EXEC
        atomicAdd(&poolAcc[64 + n], s1);           //  stays uniform around WMMA loop)
    }
    __syncthreads();
    // flush: one global atomic per (image, channel) per block
    int slot = threadIdx.x >> 6, n = threadIdx.x & 63;
    if (slot == 0 || bLast != bFirst) {
        atomicAdd(&pool[(bFirst + slot) * 64 + n],
                  poolAcc[threadIdx.x] * (1.0f / 784.0f));
    }
}

// ---------------- kernel 6: FC + log_softmax ----------------
__global__ void head_kernel(const float* __restrict__ pool, const float* __restrict__ wc,
                            const float* __restrict__ bc, float* __restrict__ out) {
    int b = blockIdx.x * blockDim.x + threadIdx.x;
    if (b >= BATCH) return;
    float logit[10];
    float mx = -1e30f;
#pragma unroll
    for (int j = 0; j < 10; ++j) {
        float a = bc[j];
        for (int c = 0; c < 64; ++c) a += pool[b * 64 + c] * wc[j * 64 + c];
        logit[j] = a;
        mx = fmaxf(mx, a);
    }
    float s = 0.f;
#pragma unroll
    for (int j = 0; j < 10; ++j) s += expf(logit[j] - mx);
    float lse = mx + logf(s);
#pragma unroll
    for (int j = 0; j < 10; ++j) out[b * 10 + j] = logit[j] - lse;
}

// ---------------- launcher ----------------
extern "C" void kernel_launch(void* const* d_in, const int* in_sizes, int n_in,
                              void* d_out, int out_size, void* d_ws, size_t ws_size,
                              hipStream_t stream) {
    const float* x  = (const float*)d_in[0];
    const float* w1 = (const float*)d_in[1];
    const float* g1 = (const float*)d_in[2];
    const float* b1 = (const float*)d_in[3];
    const float* m1 = (const float*)d_in[4];
    const float* v1 = (const float*)d_in[5];
    const float* wp = (const float*)d_in[6];
    const float* bp = (const float*)d_in[7];
    const float* w2 = (const float*)d_in[8];
    const float* g2 = (const float*)d_in[9];
    const float* b2 = (const float*)d_in[10];
    const float* m2 = (const float*)d_in[11];
    const float* v2 = (const float*)d_in[12];
    const float* w3 = (const float*)d_in[13];
    const float* g3 = (const float*)d_in[14];
    const float* b3 = (const float*)d_in[15];
    const float* m3 = (const float*)d_in[16];
    const float* v3 = (const float*)d_in[17];
    const float* wc = (const float*)d_in[18];
    const float* bc = (const float*)d_in[19];
    float* out = (float*)d_out;

    float* ws = (float*)d_ws;
    float* h1   = ws + WS_H1;
    float* offs = ws + WS_OFF;
    float* h2   = ws + WS_H2;
    float* pool = ws + WS_POOL;
    unsigned short* wb  = (unsigned short*)(ws + WS_WB);
    unsigned short* wpb = wb;
    unsigned short* w2b = wb + 32 * KDIM;
    unsigned short* w3b = wb + 64 * KDIM;

    zero_pool_kernel<<<64, 256, 0, stream>>>(pool);
    conv1_kernel<<<(BATCH * 32 * HW + 255) / 256, 256, 0, stream>>>(
        x, w1, g1, b1, m1, v1, h1);
    convert_weights_kernel<<<(128 * KDIM + 255) / 256, 256, 0, stream>>>(wp, w2, w3, wb);
    pconv_wmma_kernel<<<MTOT / 64, 128, 0, stream>>>(h1, wpb, bp, offs);
    deform_wmma_kernel<<<MTOT / 64, 128, 0, stream>>>(h1, offs, w2b, g2, b2, m2, v2, h2);
    conv3_wmma_kernel<<<MTOT / 64, 128, 0, stream>>>(h2, w3b, g3, b3, m3, v3, pool);
    head_kernel<<<1, 256, 0, stream>>>(pool, wc, bc, out);
}
